// Model_75952201662888
// MI455X (gfx1250) — compile-verified, matrix operand order checked by reading, not run on hardware
//
#include <hip/hip_runtime.h>

// ---------------------------------------------------------------------------
// MI455X (gfx1250) implementation.
// All heavy ops are row-wise time GEMMs lowered to v_wmma_f32_16x16x32_f16:
//   rfft/irfft -> dense DFT-basis GEMMs (bases built on device)
//   gated MLP  -> fused dual-WMMA (value * gelu(gate))
//   irfft1 @ fc_idp -> single precomputed 194x96 GEMM (saves a 200MB pass)
// Residual adds and (1+gate) scalings are folded into GEMM A-staging, so no
// standalone elementwise passes touch HBM.  Each wave owns two 16-row slabs
// so every B fragment feeds two WMMAs (1.5 LDS fragment loads per WMMA).
// ---------------------------------------------------------------------------

typedef __attribute__((ext_vector_type(16))) _Float16 v16h;
typedef __attribute__((ext_vector_type(8)))  _Float16 v8h;
typedef __attribute__((ext_vector_type(8)))  float    v8f;

#define ROWS   262144L   // B*C*N = 16*32*512

__device__ __forceinline__ float gelu_f(float x) {
  const float k0 = 0.7978845608028654f, k1 = 0.044715f;
  return 0.5f * x * (1.0f + tanhf(k0 * (x + k1 * x * x * x)));
}

// Load a 16-half WMMA fragment from LDS: 8 halves at +0, 8 at +16 (two b128s).
__device__ __forceinline__ v16h frag_ld(const _Float16* p) {
  v8h lo = *(const v8h*)p;
  v8h hi = *(const v8h*)(p + 16);
  v16h r;
#pragma unroll
  for (int i = 0; i < 8; ++i) { r[i] = lo[i]; r[8 + i] = hi[i]; }
  return r;
}

__device__ __forceinline__ v8f wmma_step(v16h a, v16h b, v8f c) {
  return __builtin_amdgcn_wmma_f32_16x16x32_f16(false, a, false, b,
                                                (short)0, c, false, false);
}

// ---------------------------------------------------------------------------
// Generic rows x (Tin -> Tout) GEMM over 256-row blocks (8 waves x 2 slabs).
//   A-staging:  a = A[r,k] * (gate ? 1+gate[b,k] : 1) + (Aadd ? Aadd[r,k] : 0)
//   EPI: 0 plain, 1 +extra[row,col], 2 +extra[col] (bias),
//        3 dual-B fused gmlp: acc0 * gelu(acc1).
// ---------------------------------------------------------------------------
template <int EPI>
__global__ void __launch_bounds__(256) gemm_rows(
    const float* __restrict__ A, int lda,
    const float* __restrict__ Aadd,     // optional residual addend (same lda)
    const float* __restrict__ gate,     // optional per-(batch,k) gate on A
    const float* __restrict__ Bg0, const float* __restrict__ Bg1,
    int Tin, int Tout,
    float* __restrict__ Out, int ldo,
    const float* __restrict__ extra, int lde)
{
  extern __shared__ _Float16 sm[];
  const int tid  = threadIdx.x;
  const int lane = tid & 31;
  const int wid  = tid >> 5;
  const int KP = (Tin + 31) & ~31;   // K padded to 32
  const int SA = KP + 8;             // LDS stride (halves): +16B kills bank conflicts
  const int NT = (Tout + 15) >> 4;
  const int NP = NT << 4;
  _Float16* Al  = sm;
  _Float16* Bl0 = Al + 256 * SA;
  _Float16* Bl1 = (EPI == 3) ? (Bl0 + NP * SA) : Bl0;
  const long row0 = (long)blockIdx.x * 256;

  // Stage A tile (f32 global -> f16 LDS) with fused gate / residual add.
  for (int r = wid; r < 256; r += 8) {
    const long row = row0 + r;
    const float* ap = A + row * lda;
    const float* xp = Aadd ? (Aadd + row * lda) : nullptr;
    const float* gp = gate ? (gate + (row >> 14) * Tin) : nullptr;  // 2^14 rows/batch
    for (int k = lane; k < KP; k += 32) {
      float v = 0.0f;
      if (k < Tin) {
        v = ap[k];
        if (gp) v *= (1.0f + gp[k]);
        if (xp) v += xp[k];
      }
      Al[r * SA + k] = (_Float16)v;
    }
  }
  // Stage B column-major (Bl[n][k]) so fragment reads are contiguous in k.
  for (int n = wid; n < NP; n += 8) {
    for (int k = lane; k < KP; k += 32) {
      float v0 = (k < Tin && n < Tout) ? Bg0[k * Tout + n] : 0.0f;
      Bl0[n * SA + k] = (_Float16)v0;
      if (EPI == 3) {
        float v1 = (k < Tin && n < Tout) ? Bg1[k * Tout + n] : 0.0f;
        Bl1[n * SA + k] = (_Float16)v1;
      }
    }
  }
  __syncthreads();

  const int mrow = lane & 15;
  const int half = (lane & 16) ? 8 : 0;   // ISA 16-bit A/B lane split
  const int m0   = wid * 32;              // two 16-row slabs per wave
  const _Float16* Abase0 = Al + (m0 + mrow) * SA + half;
  const _Float16* Abase1 = Abase0 + 16 * SA;

  for (int nt = 0; nt < NT; ++nt) {
    v8f acc0 = {}, acc1 = {};
    v8f acq0 = {}, acq1 = {};
    const _Float16* Bbase  = Bl0 + (nt * 16 + mrow) * SA + half;
    const _Float16* Bbase2 = Bl1 + (nt * 16 + mrow) * SA + half;
    for (int ks = 0; ks < KP; ks += 32) {
      v16h b  = frag_ld(Bbase + ks);
      v16h a0 = frag_ld(Abase0 + ks);
      v16h a1 = frag_ld(Abase1 + ks);
      acc0 = wmma_step(a0, b, acc0);
      acc1 = wmma_step(a1, b, acc1);
      if (EPI == 3) {
        v16h b2 = frag_ld(Bbase2 + ks);
        acq0 = wmma_step(a0, b2, acq0);
        acq1 = wmma_step(a1, b2, acq1);
      }
    }
    const int col = nt * 16 + mrow;
    if (col < Tout) {
      const int mofs = (lane & 16) ? 8 : 0;  // D: VGPR v -> M=v (+8 hi lanes)
#pragma unroll
      for (int s = 0; s < 2; ++s) {
        v8f acc = s ? acc1 : acc0;
        v8f acq = s ? acq1 : acq0;
#pragma unroll
        for (int v = 0; v < 8; ++v) {
          long row = row0 + m0 + s * 16 + mofs + v;
          float val = acc[v];
          if (EPI == 1) val += extra[row * lde + col];
          else if (EPI == 2) val += extra[col];
          else if (EPI == 3) val *= gelu_f(acq[v]);
          Out[row * ldo + col] = val;
        }
      }
    }
  }
}

// ---------------------------------------------------------------------------
// Per-(b,n) cross-channel spectral mix: re' = Lc[l,h] * sum_c Re[c,h]*Gc[c,l]
// (imag via Gt/Lt).  S layout: [b,c,n][0..96]=Re, [97..193]=Im, ld=194.
// ---------------------------------------------------------------------------
__global__ void __launch_bounds__(256) mix_kernel(
    const float* __restrict__ S, float* __restrict__ M,
    const float* __restrict__ Gc, const float* __restrict__ Lc,
    const float* __restrict__ Gt, const float* __restrict__ Lt)
{
  __shared__ float Sm[32][194];
  __shared__ float Gcs[32][32], Gts[32][32];
  __shared__ float Lcs[32][97], Lts[32][97];
  const int tid = threadIdx.x, lane = tid & 31, wid = tid >> 5;
  const int bn = blockIdx.x;
  const int b = bn >> 9, n = bn & 511;
  for (int c = wid; c < 32; c += 8) {
    const float* sp = S + (((long)b * 32 + c) * 512 + n) * 194;
    for (int h = lane; h < 194; h += 32) Sm[c][h] = sp[h];
  }
  for (int i = tid; i < 32 * 32; i += 256) { Gcs[i >> 5][i & 31] = Gc[i]; Gts[i >> 5][i & 31] = Gt[i]; }
  for (int i = tid; i < 32 * 97; i += 256) { int l = i / 97, h = i % 97; Lcs[l][h] = Lc[i]; Lts[l][h] = Lt[i]; }
  __syncthreads();
  for (int idx = tid; idx < 32 * 194; idx += 256) {
    int l = idx / 194, h = idx % 194;
    float acc = 0.f, val;
    if (h < 97) {
      for (int c = 0; c < 32; ++c) acc += Sm[c][h] * Gcs[c][l];
      val = Lcs[l][h] * acc;
    } else {
      int hh = h - 97;
      for (int c = 0; c < 32; ++c) acc += Sm[c][h] * Gts[c][l];
      val = Lts[l][hh] * acc;
    }
    M[(((long)b * 32 + l) * 512 + n) * 194 + h] = val;
  }
}

// mean over n: feats[b,t,c] = mean_n Z[(b,c,n),t]
__global__ void pool_kernel(const float* __restrict__ Z, int T,
                            float* __restrict__ feats)
{
  int bc = blockIdx.x;                 // b*32 + c
  int b = bc >> 5, c = bc & 31;
  long base = (long)bc * 512 * T;
  int t = threadIdx.x;
  if (t < T) {
    float acc = 0.f;
    for (int n = 0; n < 512; ++n) acc += Z[base + (long)n * T + t];
    feats[((long)b * T + t) * 32 + c] = acc * (1.0f / 512.0f);
  }
}

// softmax-diagonal attention gate: g[b,t] = gelu(softmax(QK^T/8)[t,t])
__global__ void __launch_bounds__(256) gate_kernel(
    const float* __restrict__ feats, const float* __restrict__ Wq,
    const float* __restrict__ Wk, int T, float* __restrict__ gates)
{
  extern __shared__ float fsm[];
  float* F   = fsm;            // T*32
  float* Q   = F + T * 32;     // T*64
  float* K   = Q + T * 64;     // T*64
  float* Wqs = K + T * 64;     // 2048
  float* Wks = Wqs + 2048;     // 2048
  int b = blockIdx.x, tid = threadIdx.x;
  for (int i = tid; i < T * 32; i += 256) F[i] = feats[(long)b * T * 32 + i];
  for (int i = tid; i < 2048; i += 256) { Wqs[i] = Wq[i]; Wks[i] = Wk[i]; }
  __syncthreads();
  if (tid < T) {
    for (int e = 0; e < 64; ++e) {
      float q = 0.f, k = 0.f;
      for (int c = 0; c < 32; ++c) {
        float f = F[tid * 32 + c];
        q += f * Wqs[c * 64 + e];
        k += f * Wks[c * 64 + e];
      }
      Q[tid * 64 + e] = q; K[tid * 64 + e] = k;
    }
  }
  __syncthreads();
  if (tid < T) {
    float m = -1e30f, s = 0.f, stt = 0.f;
    for (int j = 0; j < T; ++j) {
      float d = 0.f;
      for (int e = 0; e < 64; ++e) d += Q[tid * 64 + e] * K[j * 64 + e];
      d *= 0.125f;                       // 1/sqrt(64)
      if (j == tid) stt = d;
      float mn = fmaxf(m, d);
      s = s * __expf(m - mn) + __expf(d - mn);
      m = mn;
    }
    gates[b * T + tid] = gelu_f(__expf(stt - m) / s);
  }
}

// In-place per-(c,freq) complex filter on stacked Re|Im spectra (ld = 2L).
__global__ void cfilter_kernel(float* __restrict__ S, int L,
                               const float* __restrict__ Wr,
                               const float* __restrict__ Wi, long rows)
{
  long total = rows * L;
  long stride = (long)gridDim.x * blockDim.x;
  for (long idx = (long)blockIdx.x * blockDim.x + threadIdx.x; idx < total; idx += stride) {
    long r = idx / L; int h = (int)(idx - r * L);
    int c = (int)((r >> 9) & 31);
    float* p = S + r * (2 * L);
    float re = p[h], im = p[L + h];
    float wr = Wr[c * L + h], wi = Wi[c * L + h];
    p[h]     = re * wr - im * wi;
    p[L + h] = re * wi + im * wr;
  }
}

// DFT / inverse-DFT bases.  Fcat[T][2L]: cos | -sin.  Icat[2L][T]: weighted inverse.
__global__ void init_bases(float* F192, float* I192, float* F96, float* I96)
{
  const float PI2 = 6.283185307179586f;
  long total = 192L * 194 + 194L * 192 + 96L * 98 + 98L * 96;
  long stride = (long)gridDim.x * blockDim.x;
  for (long i = (long)blockIdx.x * blockDim.x + threadIdx.x; i < total; i += stride) {
    long j = i;
    if (j < 192L * 194) {
      int t = (int)(j / 194), h = (int)(j % 194);
      F192[j] = (h < 97) ? cosf(PI2 * t * h / 192.0f)
                         : -sinf(PI2 * t * (h - 97) / 192.0f);
      continue;
    }
    j -= 192L * 194;
    if (j < 194L * 192) {
      int h = (int)(j / 192), t = (int)(j % 192);
      if (h < 97) {
        float w = (h == 0 || h == 96) ? 1.0f : 2.0f;
        I192[j] = w * cosf(PI2 * t * h / 192.0f) / 192.0f;
      } else {
        int hh = h - 97;
        float w = (hh == 0 || hh == 96) ? 1.0f : 2.0f;
        I192[j] = -w * sinf(PI2 * t * hh / 192.0f) / 192.0f;
      }
      continue;
    }
    j -= 194L * 192;
    if (j < 96L * 98) {
      int t = (int)(j / 98), h = (int)(j % 98);
      F96[j] = (h < 49) ? cosf(PI2 * t * h / 96.0f)
                        : -sinf(PI2 * t * (h - 49) / 96.0f);
      continue;
    }
    j -= 96L * 98;
    {
      int h = (int)(j / 96), t = (int)(j % 96);
      if (h < 49) {
        float w = (h == 0 || h == 48) ? 1.0f : 2.0f;
        I96[j] = w * cosf(PI2 * t * h / 96.0f) / 96.0f;
      } else {
        int hh = h - 49;
        float w = (hh == 0 || hh == 48) ? 1.0f : 2.0f;
        I96[j] = -w * sinf(PI2 * t * hh / 96.0f) / 96.0f;
      }
    }
  }
}

// C2 = IDFT192(194x192) @ Wfc(192x96): fuses irfft1 with fc_idp.
__global__ void init_c2(const float* __restrict__ I192,
                        const float* __restrict__ Wfc, float* __restrict__ C2)
{
  int idx = blockIdx.x * blockDim.x + threadIdx.x;
  if (idx < 194 * 96) {
    int h = idx / 96, s = idx % 96;
    float acc = 0.f;
    for (int t = 0; t < 192; ++t) acc += I192[h * 192 + t] * Wfc[t * 96 + s];
    C2[idx] = acc;
  }
}

static size_t gemm_smem(int Tin, int Tout, bool dual) {
  int KP = (Tin + 31) & ~31, SA = KP + 8;
  int NP = ((Tout + 15) / 16) * 16;
  return ((size_t)256 * SA + (size_t)NP * SA * (dual ? 2 : 1)) * sizeof(_Float16);
}

extern "C" void kernel_launch(void* const* d_in, const int* in_sizes, int n_in,
                              void* d_out, int out_size, void* d_ws, size_t ws_size,
                              hipStream_t stream) {
  (void)in_sizes; (void)n_in; (void)out_size; (void)ws_size;
  const float* x   = (const float*)d_in[0];
  const float* Gc  = (const float*)d_in[3];
  const float* Lc  = (const float*)d_in[4];
  const float* Gt  = (const float*)d_in[5];
  const float* Lt  = (const float*)d_in[6];
  const float* f1r = (const float*)d_in[7];
  const float* f1i = (const float*)d_in[8];
  const float* f2r = (const float*)d_in[9];
  const float* f2i = (const float*)d_in[10];
  const float* Wfc = (const float*)d_in[11];
  const float* bfc = (const float*)d_in[12];
  const float* mi1 = (const float*)d_in[13];
  const float* mi2 = (const float*)d_in[14];
  const float* mo1 = (const float*)d_in[15];
  const float* mo2 = (const float*)d_in[16];
  const float* aiq = (const float*)d_in[17];
  const float* aik = (const float*)d_in[18];
  const float* aoq = (const float*)d_in[19];
  const float* aok = (const float*)d_in[20];
  float* out = (float*)d_out;
  float* ws  = (float*)d_ws;

  // Workspace: two rows*194 buffers, one rows*96, small tail for bases/feats.
  float* W0 = ws;
  float* W1 = ws + ROWS * 194;
  float* W2 = ws + 2 * ROWS * 194;
  float* F192 = W2 + ROWS * 96;
  float* I192 = F192 + 192 * 194;
  float* F96  = I192 + 194 * 192;
  float* I96  = F96 + 96 * 98;
  float* C2   = I96 + 98 * 96;
  float* feats = C2 + 194 * 96;
  float* gates = feats + 16 * 192 * 32;

  init_bases<<<128, 256, 0, stream>>>(F192, I192, F96, I96);
  init_c2<<<(194 * 96 + 255) / 256, 256, 0, stream>>>(I192, Wfc, C2);

  // S = rfft(x)  (rows x 192 @ 192 x 194)
  gemm_rows<0><<<1024, 256, gemm_smem(192, 194, false), stream>>>(
      x, 192, nullptr, nullptr, F192, nullptr, 192, 194, W1, 194, nullptr, 0);
  // per-frequency channel mix
  mix_kernel<<<16 * 512, 256, 0, stream>>>(W1, W0, Gc, Lc, Gt, Lt);
  // x_c = irfft(mixed) + x
  gemm_rows<1><<<1024, 256, gemm_smem(194, 192, false), stream>>>(
      W0, 194, nullptr, nullptr, I192, nullptr, 194, 192, W1, 192, x, 192);

  // input gated-MLP + attention-gate layers; gate applied lazily at next use.
  // layer 0: z1 = gmlp(x_c)            (W1 -> W0), gate g1 from raw z1
  gemm_rows<3><<<1024, 256, gemm_smem(192, 192, true), stream>>>(
      W1, 192, nullptr, nullptr, mi1, mi2, 192, 192, W0, 192, nullptr, 0);
  pool_kernel<<<512, 256, 0, stream>>>(W0, 192, feats);
  gate_kernel<<<16, 256, (size_t)(192 * 32 + 2 * 192 * 64 + 4096) * 4, stream>>>(
      feats, aiq, aik, 192, gates);
  // layer 1: z2 = gmlp(z1*(1+g1))      (W0 -> W1), gate g2 from raw z2
  gemm_rows<3><<<1024, 256, gemm_smem(192, 192, true), stream>>>(
      W0, 192, nullptr, gates, mi1 + 192 * 192, mi2 + 192 * 192,
      192, 192, W1, 192, nullptr, 0);
  pool_kernel<<<512, 256, 0, stream>>>(W1, 192, feats);
  gate_kernel<<<16, 256, (size_t)(192 * 32 + 2 * 192 * 64 + 4096) * 4, stream>>>(
      feats, aiq + 2048, aik + 2048, 192, gates);

  // fconv1: S2 = rfft(x + z2*(1+g2)) -- residual add + gate fused into staging
  gemm_rows<0><<<1024, 256, gemm_smem(192, 194, false), stream>>>(
      W1, 192, x, gates, F192, nullptr, 192, 194, W0, 194, nullptr, 0);
  cfilter_kernel<<<8192, 256, 0, stream>>>(W0, 97, f1r, f1i, ROWS);
  // h_y = S2' @ (IDFT192 @ Wfc) + bias
  gemm_rows<2><<<1024, 256, gemm_smem(194, 96, false), stream>>>(
      W0, 194, nullptr, nullptr, C2, nullptr, 194, 96, W2, 96, bfc, 0);

  // output gated-MLP + gate layers (scratch carved from W0)
  float* Wp1 = W0;
  float* Wp2 = W0 + ROWS * 96;
  // layer 0: y1 = gmlp(h_y)            (W2 -> Wp1), gate go1
  gemm_rows<3><<<1024, 256, gemm_smem(96, 96, true), stream>>>(
      W2, 96, nullptr, nullptr, mo1, mo2, 96, 96, Wp1, 96, nullptr, 0);
  pool_kernel<<<512, 256, 0, stream>>>(Wp1, 96, feats);
  gate_kernel<<<16, 256, (size_t)(96 * 32 + 2 * 96 * 64 + 4096) * 4, stream>>>(
      feats, aoq, aok, 96, gates);
  // layer 1: y2 = gmlp(y1*(1+go1))     (Wp1 -> Wp2), gate go2
  gemm_rows<3><<<1024, 256, gemm_smem(96, 96, true), stream>>>(
      Wp1, 96, nullptr, gates, mo1 + 96 * 96, mo2 + 96 * 96,
      96, 96, Wp2, 96, nullptr, 0);
  pool_kernel<<<512, 256, 0, stream>>>(Wp2, 96, feats);
  gate_kernel<<<16, 256, (size_t)(96 * 32 + 2 * 96 * 64 + 4096) * 4, stream>>>(
      feats, aoq + 2048, aok + 2048, 96, gates);

  // fconv2: S3 = rfft(h_y + y2*(1+go2)); filter; irfft -> out
  float* S3 = W1;
  gemm_rows<0><<<1024, 256, gemm_smem(96, 98, false), stream>>>(
      Wp2, 96, W2, gates, F96, nullptr, 96, 98, S3, 98, nullptr, 0);
  cfilter_kernel<<<8192, 256, 0, stream>>>(S3, 49, f2r, f2i, ROWS);
  gemm_rows<0><<<1024, 256, gemm_smem(98, 96, false), stream>>>(
      S3, 98, nullptr, nullptr, I96, nullptr, 98, 96, out, 96, nullptr, 0);
}
